// GRUCellTaskSpecific_20478404067888
// MI455X (gfx1250) — matrix operand order repaired and checked by reading, not compiled, not run on hardware
//
#include <hip/hip_runtime.h>
#include <cstdint>

// ---------------------------------------------------------------------------
// GRU cell (3-source) : bf16 staging pass -> single fused WMMA GEMM kernel
// (4 accumulator groups z / r / n_xs / n_h) + gate epilogue.
// CDNA5 specifics: v_wmma_f32_16x16x32_bf16, global_load_async_to_lds_b128
// double-buffered pipeline (one barrier per K-step), XOR-swizzled LDS
// (bank-conflict free, no padding).
// ---------------------------------------------------------------------------

typedef __attribute__((ext_vector_type(16))) __bf16 v16bf;
typedef __attribute__((ext_vector_type(8)))  float  v8f;

#define BATCH 16384
#define HDIM  1024
#define KDIM  1024        // K per source (D_IN == H == HS == 1024)
#define BM    128
#define BN    64
#define BK    32

// LDS buffer layout (bytes), per stage buffer:
//   A region: 3 sources x 128 rows x 64B (32 bf16)        -> 24576 B
//   W region: 9 (src,gate) x 64 N-rows x 64B              -> 36864 B
#define ABYTES   24576u
#define BUFBYTES 61440u   // ABYTES + 36864

// f32 -> bf16 round-to-nearest-even (finite inputs)
__device__ __forceinline__ unsigned short f2bf(float f) {
  union { float f; unsigned u; } v; v.f = f;
  unsigned r = v.u + 0x7FFFu + ((v.u >> 16) & 1u);
  return (unsigned short)(r >> 16);
}

__device__ __forceinline__ float sigmoidf_(float x) {
  return 1.0f / (1.0f + __expf(-x));
}

// CDNA5 async global->LDS copy, 16B per lane, tracked by ASYNCcnt.
__device__ __forceinline__ void async_b128(unsigned lds_addr, const void* gptr) {
  asm volatile("global_load_async_to_lds_b128 %0, %1, off"
               :: "v"(lds_addr), "v"((unsigned long long)(uintptr_t)gptr)
               : "memory");
}
__device__ __forceinline__ void wait_async0() {
  asm volatile("s_wait_asynccnt 0" ::: "memory");
}

// --------------------- conversion: f32 -> bf16, 8 elems/thread -------------
__global__ void cvt_bf16_kernel(const float* __restrict__ in,
                                unsigned short* __restrict__ out, int n8) {
  int i = blockIdx.x * blockDim.x + threadIdx.x;
  if (i >= n8) return;
  const float4* p = (const float4*)in + 2 * (size_t)i;
  float4 a = p[0], b = p[1];
  union { unsigned short s[8]; uint4 q; } o;
  o.s[0] = f2bf(a.x); o.s[1] = f2bf(a.y); o.s[2] = f2bf(a.z); o.s[3] = f2bf(a.w);
  o.s[4] = f2bf(b.x); o.s[5] = f2bf(b.y); o.s[6] = f2bf(b.z); o.s[7] = f2bf(b.w);
  ((uint4*)out)[i] = o.q;
}

// ------------- weight transpose+convert: W[1024][3072] -> Wt[3072][1024] ---
__global__ void cvt_w_transpose_kernel(const float* __restrict__ W,
                                       unsigned short* __restrict__ Wt) {
  __shared__ float tile[32][33];
  int bn = blockIdx.x, bk = blockIdx.y, t = threadIdx.x;
#pragma unroll
  for (int p = 0; p < 4; ++p) {
    int idx = t + 256 * p, r = idx >> 5, c = idx & 31;
    tile[r][c] = W[(size_t)(bk * 32 + r) * 3072 + bn * 32 + c];
  }
  __syncthreads();
#pragma unroll
  for (int p = 0; p < 4; ++p) {
    int idx = t + 256 * p, r = idx >> 5, c = idx & 31;
    Wt[(size_t)(bn * 32 + r) * 1024 + bk * 32 + c] = f2bf(tile[c][r]);
  }
}

// --------------------------- fused GEMM + GRU gate -------------------------
union FragBF { uint4 q[2]; v16bf v; };

__global__ __launch_bounds__(256)
void gru_wmma_kernel(const unsigned short* __restrict__ Xb,
                     const unsigned short* __restrict__ Hb,
                     const unsigned short* __restrict__ Sb,
                     const unsigned short* __restrict__ Wxt,   // [3072][1024] N-major
                     const unsigned short* __restrict__ Wht,
                     const unsigned short* __restrict__ Wst,
                     const float* __restrict__ hprev,
                     const float* __restrict__ bx,
                     const float* __restrict__ bh,
                     const float* __restrict__ bs,
                     float* __restrict__ out) {
  __shared__ unsigned char smem[2][BUFBYTES];        // 120 KB double buffer

  const int tileM = blockIdx.y * BM;
  const int tileN = blockIdx.x * BN;                 // h-column base

  const int tid  = threadIdx.x;
  const int wave = tid >> 5;
  const int lane = tid & 31;
  const int wm   = wave >> 2;                        // 0..1 : 64-row slab
  const int wn   = wave & 3;                         // 0..3 : 16-col slab
  const int lm   = lane & 15;
  const int lh   = lane >> 4;

  // XOR swizzle: k-chunk c (16B units) of row r lives at c ^ ((r>>2)&3).
  // For all fragment reads the row term reduces to (lm>>2)&3 (lane const).
  const int      sw    = (lm >> 2) & 3;
  const unsigned aoff0 = (unsigned)((lh ^ sw) * 16);          // K chunk lh
  const unsigned aoff1 = (unsigned)(((lh + 2) ^ sw) * 16);    // K chunk lh+2
  const unsigned boff0 = (unsigned)((lh ^ sw) * 16);          // K chunk lh
  const unsigned boff1 = (unsigned)(((lh + 2) ^ sw) * 16);
  const unsigned abase = (unsigned)((wm * 64 + lm) * 64);     // + i*1024 imm
  const unsigned bbase = ABYTES + (unsigned)((wn * 16 + lm) * 64); // + sg*4096

  // Per-thread staging coordinates (16B chunk each): row = tid>>2, c4 = tid&3.
  const int      srow  = tid >> 2;                   // 0..63
  const int      sc4   = tid & 3;
  const unsigned scpos = (unsigned)((sc4 ^ ((srow >> 2) & 3)) * 16);
  const unsigned smem0 = (unsigned)(uintptr_t)&smem[0][0];  // LDS byte address

  // accumulators: [gate][ifrag]; gates 0=z 1=r 2=n_x+n_s 3=n_h
  v8f acc[4][4];
#pragma unroll
  for (int g = 0; g < 4; ++g)
#pragma unroll
    for (int i = 0; i < 4; ++i) acc[g][i] = v8f{};

  // -------- async stage of one K-slab into LDS buffer `bufoff` -------------
  auto stage = [&](unsigned bufoff, int kk) {
#pragma unroll
    for (int s = 0; s < 3; ++s) {                    // A: rows srow, srow+64
      const unsigned short* ap = (s == 0) ? Xb : ((s == 1) ? Hb : Sb);
#pragma unroll
      for (int it = 0; it < 2; ++it) {
        async_b128(smem0 + bufoff + (unsigned)(s * 8192 + it * 4096) +
                       (unsigned)(srow * 64) + scpos,
                   ap + (size_t)(tileM + srow + it * 64) * KDIM + kk + sc4 * 8);
      }
    }
#pragma unroll
    for (int sg = 0; sg < 9; ++sg) {                 // W: N-row srow
      const int s = sg / 3, g = sg - 3 * s;          // compile-time
      const unsigned short* wp = (s == 0) ? Wxt : ((s == 1) ? Wht : Wst);
      async_b128(smem0 + bufoff + ABYTES + (unsigned)(sg * 4096) +
                     (unsigned)(srow * 64) + scpos,
                 wp + (size_t)(g * HDIM + tileN + srow) * KDIM + kk + sc4 * 8);
    }
  };

  stage(0u, 0);

  for (int kk = 0; kk < KDIM; kk += BK) {
    const unsigned curbuf = (unsigned)(((kk >> 5) & 1)) * BUFBYTES;
    wait_async0();                 // this wave's stage(cur) complete
    __syncthreads();               // whole WG: LDS(cur) valid; all waves have
                                   // consumed (dscnt-waited) buf(cur^1) reads,
                                   // so staging into it below is WAR-safe.
    if (kk + BK < KDIM) stage(curbuf ^ BUFBYTES, kk + BK);  // overlap w/ math

    const unsigned char* buf = &smem[0][0] + curbuf;

#pragma unroll
    for (int s = 0; s < 3; ++s) {
      FragBF a[4];
#pragma unroll
      for (int i = 0; i < 4; ++i) {
        const unsigned char* p = buf + s * 8192 + abase + i * 1024;
        a[i].q[0] = *(const uint4*)(p + aoff0);
        a[i].q[1] = *(const uint4*)(p + aoff1);
      }
#pragma unroll
      for (int g = 0; g < 3; ++g) {
        FragBF b;
        const unsigned char* pw = buf + bbase + (s * 3 + g) * 4096;
        b.q[0] = *(const uint4*)(pw + boff0);
        b.q[1] = *(const uint4*)(pw + boff1);
        const int ga = (g == 2) ? ((s == 1) ? 3 : 2) : g;  // n splits by source
#pragma unroll
        for (int i = 0; i < 4; ++i)
          acc[ga][i] = __builtin_amdgcn_wmma_f32_16x16x32_bf16(
              false, a[i].v, false, b.v, (short)0, acc[ga][i], false, false);
      }
    }
    // no trailing barrier: next iteration's wait+barrier precedes any reuse
  }

  // ---- fused GRU gate epilogue (all four accs share C/D fragment layout) --
  const int nh = tileN + wn * 16 + lm;
  const float Bz  = bx[nh] + bh[nh] + bs[nh];
  const float Br  = bx[HDIM + nh] + bh[HDIM + nh] + bs[HDIM + nh];
  const float Bnx = bx[2 * HDIM + nh] + bs[2 * HDIM + nh];
  const float Bnh = bh[2 * HDIM + nh];

#pragma unroll
  for (int i = 0; i < 4; ++i) {
#pragma unroll
    for (int vi = 0; vi < 8; ++vi) {
      const int m = tileM + wm * 64 + i * 16 + vi + lh * 8;  // C/D row map
      const float z  = sigmoidf_(acc[0][i][vi] + Bz);
      const float r  = sigmoidf_(acc[1][i][vi] + Br);
      const float nm = tanhf(acc[2][i][vi] + Bnx + r * (acc[3][i][vi] + Bnh));
      const float hp = hprev[(size_t)m * HDIM + nh];
      out[(size_t)m * HDIM + nh] = (1.0f - z) * nm + z * hp;
    }
  }
}

// ---------------------------------------------------------------------------
extern "C" void kernel_launch(void* const* d_in, const int* in_sizes, int n_in,
                              void* d_out, int out_size, void* d_ws, size_t ws_size,
                              hipStream_t stream) {
  const float* x     = (const float*)d_in[0];
  const float* hprev = (const float*)d_in[1];
  const float* hs    = (const float*)d_in[2];
  const float* Wx    = (const float*)d_in[3];
  const float* Wh    = (const float*)d_in[4];
  const float* Ws    = (const float*)d_in[5];
  const float* bx    = (const float*)d_in[6];
  const float* bh    = (const float*)d_in[7];
  const float* bs    = (const float*)d_in[8];
  float* out = (float*)d_out;

  // bf16 staging layout in workspace (~114 MiB total; fits 192 MB L2)
  unsigned short* wsp = (unsigned short*)d_ws;
  const size_t ACT = (size_t)BATCH * KDIM;
  const size_t WEL = (size_t)KDIM * 3 * HDIM;
  unsigned short* Xb  = wsp;
  unsigned short* Hb  = Xb + ACT;
  unsigned short* Sb  = Hb + ACT;
  unsigned short* Wxt = Sb + ACT;
  unsigned short* Wht = Wxt + WEL;
  unsigned short* Wst = Wht + WEL;

  {
    int n8 = (int)(ACT / 8);
    int blocks = (n8 + 255) / 256;
    cvt_bf16_kernel<<<blocks, 256, 0, stream>>>(x,     Xb, n8);
    cvt_bf16_kernel<<<blocks, 256, 0, stream>>>(hprev, Hb, n8);
    cvt_bf16_kernel<<<blocks, 256, 0, stream>>>(hs,    Sb, n8);
  }
  {
    dim3 tg(3 * HDIM / 32, KDIM / 32);  // (96, 32)
    cvt_w_transpose_kernel<<<tg, 256, 0, stream>>>(Wx, Wxt);
    cvt_w_transpose_kernel<<<tg, 256, 0, stream>>>(Wh, Wht);
    cvt_w_transpose_kernel<<<tg, 256, 0, stream>>>(Ws, Wst);
  }
  {
    dim3 grid(HDIM / BN, BATCH / BM);   // (16, 128)
    gru_wmma_kernel<<<grid, 256, 0, stream>>>(Xb, Hb, Sb, Wxt, Wht, Wst,
                                              hprev, bx, bh, bs, out);
  }
}
